// FactoredAttention_16733192585278
// MI455X (gfx1250) — matrix operand order repaired and checked by reading, not compiled, not run on hardware
//
#include <hip/hip_runtime.h>
#include <hip/hip_bf16.h>

// FactoredAttention on MI455X (gfx1250, wave32, WMMA).
// Pipeline: QKV GEMM (f16 WMMA) -> 8192 x [64-seq, 64-hd] causal attention
// (f16 WMMA) -> output GEMM (f16 WMMA, f32 out).

typedef __attribute__((ext_vector_type(16))) _Float16 v16h;
typedef __attribute__((ext_vector_type(8)))  float    v8f;
typedef decltype(__builtin_amdgcn_cvt_pkrtz(0.0f, 0.0f)) cvt_h2;  // v2f16

#define BATCH 4
#define SEQ   8192
#define DMODEL 1024
#define NHEAD 16
#define BLKL  128
#define HDIM  64
#define NBLK  64
#define NPROB (BATCH * BLKL * NHEAD)   // 8192 attention problems
#define THREED 3072

#define MT 128
#define NT 128
#define KT 32
#define LSTR 40   // LDS row stride in halves: 16B-aligned rows, conflict-skewed

// ---------------------------------------------------------------------------
// WMMA fragment loaders (per ISA 7.12.2 VGPR layouts, wave32).
// A (16x32 f16, MxK): lane holds M=lane&15; dword j covers K pair
//   {2j (j<4) | 2j+8 (j>=4)} + 8*(lane>=16).
// B (32x16 f16, KxN): lane holds N=lane&15; dword j covers K pair
//   {2j (j<4) | 2j+8 (j>=4)} + 8*(lane>=16).
// Both read from a row-major [entity][k] array; rowp points at [row][k0].
// ---------------------------------------------------------------------------
__device__ __forceinline__ v16h load_frag_a(const _Float16* rowp, int lane) {
  union { v16h h; unsigned u[8]; } f;
  const int khi = (lane & 16) >> 1;  // +8 for upper half-wave
#pragma unroll
  for (int j = 0; j < 8; ++j) {
    const int k = ((j < 4) ? (2 * j) : (2 * j + 8)) + khi;
    f.u[j] = *(const unsigned*)(rowp + k);
  }
  return f.h;
}

__device__ __forceinline__ v16h load_frag_b(const _Float16* rowp, int lane) {
  union { v16h h; unsigned u[8]; } f;
  const int khi = (lane & 16) >> 1;  // +8 for upper half-wave
#pragma unroll
  for (int j = 0; j < 8; ++j) {
    f.u[j] = *(const unsigned*)(rowp + ((j < 4) ? (2 * j) : (2 * j + 8)) + khi);
  }
  return f.h;
}

// Pack 8 f32 -> 8 f16 (v_cvt_pk_rtz_f16_f32 x4) as one 16-byte value.
__device__ __forceinline__ uint4 pack8_f16(float4 a, float4 b) {
  union { cvt_h2 h[4]; uint4 u; } p;
  p.h[0] = __builtin_amdgcn_cvt_pkrtz(a.x, a.y);
  p.h[1] = __builtin_amdgcn_cvt_pkrtz(a.z, a.w);
  p.h[2] = __builtin_amdgcn_cvt_pkrtz(b.x, b.y);
  p.h[3] = __builtin_amdgcn_cvt_pkrtz(b.z, b.w);
  return p.u;
}

// ---------------------------------------------------------------------------
// Kernel 1: QKV projection. C[m,n] = sum_k X[m,k]*Wqkv[n,k] + bqkv[n]
// M = B*S = 32768, N = 3*D = 3072, K = D = 1024.
// Software-pipelined: tile k+1's b128 global loads are issued before tile k's
// WMMA block, overlapping HBM latency with matrix math.
// Epilogue scatters f16 results into attention layouts:
//   Q,K : [p=(b*128+bl)*16+h][nb][hd]      (64x64 per problem)
//   Vt  : [p][hd][nb]                      (pre-transposed for the PV WMMA)
// ---------------------------------------------------------------------------
__global__ __launch_bounds__(256) void qkv_gemm_f16(
    const float* __restrict__ X, const float* __restrict__ Wqkv,
    const float* __restrict__ bqkv,
    _Float16* __restrict__ Qb, _Float16* __restrict__ Kb,
    _Float16* __restrict__ Vt) {
  __shared__ __align__(16) _Float16 As[MT * LSTR];
  __shared__ __align__(16) _Float16 Bs[NT * LSTR];

  const int tid  = threadIdx.x;
  const int lane = tid & 31;
  const int wave = tid >> 5;
  const int m0 = blockIdx.y * MT;
  const int n0 = blockIdx.x * NT;
  const int mw = (wave & 3) * 32;   // 2 M-tiles per wave
  const int nw = (wave >> 2) * 64;  // 4 N-tiles per wave

  // staging geometry: 512 chunks of 8 floats per 128x32 tile, 2 per thread
  const int c0r = (0 * 256 + tid) >> 2, c0c = ((0 * 256 + tid) & 3) * 8;
  const int c1r = (1 * 256 + tid) >> 2, c1c = ((1 * 256 + tid) & 3) * 8;

  float4 ra[2][2], rb[2][2];
  v8f acc[2][4] = {};

  auto load_tile = [&](int kt) {
    const float* pa0 = X    + (size_t)(m0 + c0r) * DMODEL + kt + c0c;
    const float* pa1 = X    + (size_t)(m0 + c1r) * DMODEL + kt + c1c;
    const float* pb0 = Wqkv + (size_t)(n0 + c0r) * DMODEL + kt + c0c;
    const float* pb1 = Wqkv + (size_t)(n0 + c1r) * DMODEL + kt + c1c;
    ra[0][0] = *(const float4*)(pa0); ra[0][1] = *(const float4*)(pa0 + 4);
    ra[1][0] = *(const float4*)(pa1); ra[1][1] = *(const float4*)(pa1 + 4);
    rb[0][0] = *(const float4*)(pb0); rb[0][1] = *(const float4*)(pb0 + 4);
    rb[1][0] = *(const float4*)(pb1); rb[1][1] = *(const float4*)(pb1 + 4);
  };
  auto store_tile = [&]() {
    *(uint4*)(As + c0r * LSTR + c0c) = pack8_f16(ra[0][0], ra[0][1]);
    *(uint4*)(As + c1r * LSTR + c1c) = pack8_f16(ra[1][0], ra[1][1]);
    *(uint4*)(Bs + c0r * LSTR + c0c) = pack8_f16(rb[0][0], rb[0][1]);
    *(uint4*)(Bs + c1r * LSTR + c1c) = pack8_f16(rb[1][0], rb[1][1]);
  };

  load_tile(0);
  for (int kt = 0; kt < DMODEL; kt += KT) {
    store_tile();
    __syncthreads();
    if (kt + KT < DMODEL) load_tile(kt + KT);   // overlap with WMMA below

    v16h a[2], b[4];
#pragma unroll
    for (int i = 0; i < 2; ++i)
      a[i] = load_frag_a(As + (mw + i * 16 + (lane & 15)) * LSTR, lane);
#pragma unroll
    for (int j = 0; j < 4; ++j)
      b[j] = load_frag_b(Bs + (nw + j * 16 + (lane & 15)) * LSTR, lane);
#pragma unroll
    for (int i = 0; i < 2; ++i)
#pragma unroll
      for (int j = 0; j < 4; ++j)
        acc[i][j] = __builtin_amdgcn_wmma_f32_16x16x32_f16(
            false, a[i], false, b[j], (short)0, acc[i][j], false, false);
    __syncthreads();
  }

  // Epilogue: C/D layout -> element r: M = r + 8*(lane>=16), N = lane&15.
  const int mrow = (lane >> 4) * 8;
  const int ncol = lane & 15;
#pragma unroll
  for (int i = 0; i < 2; ++i) {
#pragma unroll
    for (int j = 0; j < 4; ++j) {
      const int n = n0 + nw + j * 16 + ncol;
      const float bias = bqkv[n];
      const int which = n >> 10;            // 0=Q 1=K 2=V
      const int h  = (n & 1023) >> 6;
      const int hd = n & 63;
      _Float16* dst = (which == 0) ? Qb : ((which == 1) ? Kb : Vt);
#pragma unroll
      for (int r = 0; r < 8; ++r) {
        const int m  = m0 + mw + i * 16 + r + mrow;  // m = b*8192 + s
        const int b_ = m >> 13;
        const int s  = m & 8191;
        const int nb = s >> 7;
        const int bl = s & 127;
        const int p  = (b_ * BLKL + bl) * NHEAD + h;
        const float v = acc[i][j][r] + bias;
        size_t off;
        if (which == 2) off = ((size_t)p * 64 + hd) * 64 + nb;  // Vt[hd][nb]
        else            off = ((size_t)p * 64 + nb) * 64 + hd;  // [nb][hd]
        dst[off] = (_Float16)v;
      }
    }
  }
}

// ---------------------------------------------------------------------------
// Kernel 2: causal attention, one 64x64 problem per workgroup (4 waves).
// Wave w owns query rows [16w, 16w+16). S=Q*K^T (scale 1/8, tril mask over
// nb), softmax per row, O=P*V; output scattered through the reference's
// reshape/swapaxes permutation into Tmp[b][s2][d2] (f16).
// ---------------------------------------------------------------------------
__global__ __launch_bounds__(128) void attn64(
    const _Float16* __restrict__ Qb, const _Float16* __restrict__ Kb,
    const _Float16* __restrict__ Vt, _Float16* __restrict__ Tmp) {
  __shared__ __align__(16) _Float16 Qs[64 * 64];
  __shared__ __align__(16) _Float16 Ks[64 * 64];
  __shared__ __align__(16) _Float16 Vs[64 * 64];
  __shared__ __align__(16) _Float16 Ps[64 * 64];
  __shared__ __align__(16) float    Sf[64 * 64];

  const int tid  = threadIdx.x;
  const int lane = tid & 31;
  const int wave = tid >> 5;
  const int p    = blockIdx.x;
  const size_t pbase = (size_t)p * 4096;

  {  // stage Q, K, Vt (8 KB each) as uint4
    const uint4* q4 = (const uint4*)(Qb + pbase);
    const uint4* k4 = (const uint4*)(Kb + pbase);
    const uint4* v4 = (const uint4*)(Vt + pbase);
    uint4* qs = (uint4*)Qs; uint4* ks = (uint4*)Ks; uint4* vs = (uint4*)Vs;
#pragma unroll
    for (int i = 0; i < 4; ++i) {
      qs[i * 128 + tid] = q4[i * 128 + tid];
      ks[i * 128 + tid] = k4[i * 128 + tid];
      vs[i * 128 + tid] = v4[i * 128 + tid];
    }
  }
  __syncthreads();

  const int arow = wave * 16 + (lane & 15);   // A-operand row for this lane
  const int mrow = (lane >> 4) * 8;

  // ---- scores S = (1/8) Q K^T, causal over nb ----
#pragma unroll
  for (int j = 0; j < 4; ++j) {
    v8f acc = {};
#pragma unroll
    for (int kk = 0; kk < 64; kk += 32) {
      const v16h a = load_frag_a(Qs + arow * 64 + kk, lane);
      const v16h b = load_frag_b(Ks + (j * 16 + (lane & 15)) * 64 + kk, lane);
      acc = __builtin_amdgcn_wmma_f32_16x16x32_f16(false, a, false, b,
                                                   (short)0, acc, false, false);
    }
    const int kcol = j * 16 + (lane & 15);
#pragma unroll
    for (int r = 0; r < 8; ++r) {
      const int q = wave * 16 + mrow + r;
      float v = acc[r] * 0.125f;              // 1/sqrt(64)
      if (kcol > q) v = -3.0e38f;             // causal mask
      Sf[q * 64 + kcol] = v;
    }
  }
  __syncthreads();

  // ---- softmax per query row ----
  if (tid < 64) {
    float mx = -3.4e38f;
    for (int k = 0; k < 64; ++k) mx = fmaxf(mx, Sf[tid * 64 + k]);
    float sum = 0.f;
    for (int k = 0; k < 64; ++k) {
      const float e = __expf(Sf[tid * 64 + k] - mx);
      Sf[tid * 64 + k] = e;
      sum += e;
    }
    const float inv = 1.f / sum;
    for (int k = 0; k < 64; ++k) Ps[tid * 64 + k] = (_Float16)(Sf[tid * 64 + k] * inv);
  }
  __syncthreads();

  // ---- O = P V, then scatter through the reference permutation ----
  const int b_  = p >> 11;           // p = (b*128 + bl)*16 + h
  const int rem = p & 2047;
  const int bl  = rem >> 4;
  const int h   = rem & 15;
#pragma unroll
  for (int j = 0; j < 4; ++j) {
    v8f acc = {};
#pragma unroll
    for (int kk = 0; kk < 64; kk += 32) {
      const v16h a = load_frag_a(Ps + arow * 64 + kk, lane);
      const v16h b = load_frag_b(Vs + (j * 16 + (lane & 15)) * 64 + kk, lane);
      acc = __builtin_amdgcn_wmma_f32_16x16x32_f16(false, a, false, b,
                                                   (short)0, acc, false, false);
    }
    const int hd = j * 16 + (lane & 15);
#pragma unroll
    for (int r = 0; r < 8; ++r) {
      const int nb  = wave * 16 + mrow + r;
      const int d2  = (nb & 15) * 64 + hd;            // reshape scramble
      const int bl2 = (bl & 31) * 4 + (nb >> 4);
      const int nb2 = h * 4 + (bl >> 5);
      const int s2  = nb2 * BLKL + bl2;
      Tmp[((size_t)b_ * SEQ + s2) * DMODEL + d2] = (_Float16)acc[r];
    }
  }
}

// ---------------------------------------------------------------------------
// Kernel 3: output projection. Out[m,n] = sum_k Tmp[m,k]*Wout[n,k] + bout[n]
// M = 32768, N = 1024, K = 1024.  Tmp is already f16 (uint4 copy into LDS);
// Wout converted fp32->f16 on stage. Same software pipelining as kernel 1.
// ---------------------------------------------------------------------------
__global__ __launch_bounds__(256) void out_gemm_f16(
    const _Float16* __restrict__ Tmp, const float* __restrict__ Wout,
    const float* __restrict__ bout, float* __restrict__ Out) {
  __shared__ __align__(16) _Float16 As[MT * LSTR];
  __shared__ __align__(16) _Float16 Bs[NT * LSTR];

  const int tid  = threadIdx.x;
  const int lane = tid & 31;
  const int wave = tid >> 5;
  const int m0 = blockIdx.y * MT;
  const int n0 = blockIdx.x * NT;
  const int mw = (wave & 3) * 32;
  const int nw = (wave >> 2) * 64;

  const int c0r = (0 * 256 + tid) >> 2, c0c = ((0 * 256 + tid) & 3) * 8;
  const int c1r = (1 * 256 + tid) >> 2, c1c = ((1 * 256 + tid) & 3) * 8;

  uint4  ra[2];
  float4 rb[2][2];
  v8f acc[2][4] = {};

  auto load_tile = [&](int kt) {
    ra[0] = *(const uint4*)(Tmp + (size_t)(m0 + c0r) * DMODEL + kt + c0c);
    ra[1] = *(const uint4*)(Tmp + (size_t)(m0 + c1r) * DMODEL + kt + c1c);
    const float* pb0 = Wout + (size_t)(n0 + c0r) * DMODEL + kt + c0c;
    const float* pb1 = Wout + (size_t)(n0 + c1r) * DMODEL + kt + c1c;
    rb[0][0] = *(const float4*)(pb0); rb[0][1] = *(const float4*)(pb0 + 4);
    rb[1][0] = *(const float4*)(pb1); rb[1][1] = *(const float4*)(pb1 + 4);
  };
  auto store_tile = [&]() {
    *(uint4*)(As + c0r * LSTR + c0c) = ra[0];
    *(uint4*)(As + c1r * LSTR + c1c) = ra[1];
    *(uint4*)(Bs + c0r * LSTR + c0c) = pack8_f16(rb[0][0], rb[0][1]);
    *(uint4*)(Bs + c1r * LSTR + c1c) = pack8_f16(rb[1][0], rb[1][1]);
  };

  load_tile(0);
  for (int kt = 0; kt < DMODEL; kt += KT) {
    store_tile();
    __syncthreads();
    if (kt + KT < DMODEL) load_tile(kt + KT);

    v16h a[2], b[4];
#pragma unroll
    for (int i = 0; i < 2; ++i)
      a[i] = load_frag_a(As + (mw + i * 16 + (lane & 15)) * LSTR, lane);
#pragma unroll
    for (int j = 0; j < 4; ++j)
      b[j] = load_frag_b(Bs + (nw + j * 16 + (lane & 15)) * LSTR, lane);
#pragma unroll
    for (int i = 0; i < 2; ++i)
#pragma unroll
      for (int j = 0; j < 4; ++j)
        acc[i][j] = __builtin_amdgcn_wmma_f32_16x16x32_f16(
            false, a[i], false, b[j], (short)0, acc[i][j], false, false);
    __syncthreads();
  }

  const int mrow = (lane >> 4) * 8;
  const int ncol = lane & 15;
#pragma unroll
  for (int i = 0; i < 2; ++i) {
#pragma unroll
    for (int j = 0; j < 4; ++j) {
      const int n = n0 + nw + j * 16 + ncol;
      const float bias = bout[n];
#pragma unroll
      for (int r = 0; r < 8; ++r) {
        const int m = m0 + mw + i * 16 + r + mrow;
        Out[(size_t)m * DMODEL + n] = acc[i][j][r] + bias;
      }
    }
  }
}

// ---------------------------------------------------------------------------
extern "C" void kernel_launch(void* const* d_in, const int* in_sizes, int n_in,
                              void* d_out, int out_size, void* d_ws, size_t ws_size,
                              hipStream_t stream) {
  (void)in_sizes; (void)n_in; (void)out_size; (void)ws_size;
  const float* x    = (const float*)d_in[0];
  const float* Wqkv = (const float*)d_in[1];
  const float* bqkv = (const float*)d_in[2];
  const float* Wout = (const float*)d_in[3];
  const float* bout = (const float*)d_in[4];
  float* out = (float*)d_out;

  char* ws = (char*)d_ws;
  const size_t PER = (size_t)NPROB * 64 * 64 * sizeof(_Float16);  // 64 MiB
  _Float16* Qb  = (_Float16*)(ws);
  _Float16* Kb  = (_Float16*)(ws + PER);
  _Float16* Vt  = (_Float16*)(ws + 2 * PER);
  _Float16* Tmp = (_Float16*)(ws + 3 * PER);

  {  // QKV projection: M=32768 (y), N=3072 (x)
    dim3 grid(THREED / NT, (BATCH * SEQ) / MT);
    qkv_gemm_f16<<<grid, 256, 0, stream>>>(x, Wqkv, bqkv, Qb, Kb, Vt);
  }
  {  // attention: one workgroup per (b, bl, h)
    attn64<<<NPROB, 128, 0, stream>>>(Qb, Kb, Vt, Tmp);
  }
  {  // output projection: M=32768 (y), N=1024 (x)
    dim3 grid(DMODEL / NT, (BATCH * SEQ) / MT);
    out_gemm_f16<<<grid, 256, 0, stream>>>(Tmp, Wout, bout, out);
  }
}
